// CUTSEncoder_75101798137973
// MI455X (gfx1250) — matrix operand order repaired and checked by reading, not compiled, not run on hardware
//
#include <hip/hip_runtime.h>
#include <hip/hip_bf16.h>

typedef __attribute__((ext_vector_type(16))) __bf16 v16bf;
typedef __attribute__((ext_vector_type(8)))  float  v8f;
typedef __attribute__((ext_vector_type(2)))  float  v2f;

#define NPIX 524288            // 8 * 256 * 256
#define IMG_HW 65536           // 256*256
#define PW 258                 // padded width/height
#define PIMG (PW * PW)         // padded image pixels

// ---------------------------------------------------------------------------
// Pack conv weights OIHW f32 -> [Cout][tap*Cin + ci] bf16 (tap-major K)
// ---------------------------------------------------------------------------
__global__ void pack_w_kernel(const float* __restrict__ w, __hip_bfloat16* __restrict__ wp,
                              int Cin, int Cout) {
    int K = Cin * 9;
    int idx = blockIdx.x * 256 + threadIdx.x;
    if (idx >= Cout * K) return;
    int o = idx / K;
    int k = idx - o * K;
    int tap = k / Cin;
    int ci  = k - tap * Cin;
    wp[idx] = __float2bfloat16(w[(o * Cin + ci) * 9 + tap]);
}

// ---------------------------------------------------------------------------
// Zero the 1-pixel border of a padded NHWC bf16 activation buffer
// ---------------------------------------------------------------------------
__global__ void border_zero_kernel(__hip_bfloat16* __restrict__ buf, int cShift) {
    int C = 1 << cShift;
    int i = blockIdx.x * 256 + threadIdx.x;
    if (i >= 8 * 1028 * C) return;
    int c = i & (C - 1);
    int r = i >> cShift;                 // 0 .. 8*1028-1
    int b = r / 1028;
    int pos = r - b * 1028;
    int y, x;
    if (pos < 258)      { y = 0;              x = pos; }
    else if (pos < 516) { y = 257;            x = pos - 258; }
    else if (pos < 772) { y = pos - 516 + 1;  x = 0; }
    else                { y = pos - 772 + 1;  x = 257; }
    buf[(((size_t)(b * PW + y) * PW + x) << cShift) + c] = __float2bfloat16(0.f);
}

// ---------------------------------------------------------------------------
// Stage 1 direct conv: x NCHW f32 [8,3,256,256] -> conv NHWC f32 [NPIX,32]
// ---------------------------------------------------------------------------
__global__ __launch_bounds__(256)
void conv1_kernel(const float* __restrict__ x, const float* __restrict__ w1,
                  const float* __restrict__ b1, float* __restrict__ out) {
    __shared__ float wl[864];
    int t = threadIdx.x;
    for (int i = t; i < 864; i += 256) wl[i] = w1[i];
    __syncthreads();
    int c  = t & 31;
    int pi = t >> 5;
    int gp = blockIdx.x * 8 + pi;
    int b = gp >> 16, rem = gp & 65535, y = rem >> 8, xx = rem & 255;
    float acc = b1[c];
    for (int ci = 0; ci < 3; ci++)
        for (int ky = 0; ky < 3; ky++) {
            int ys = y + ky - 1;
            if ((unsigned)ys < 256u)
                for (int kx = 0; kx < 3; kx++) {
                    int xs = xx + kx - 1;
                    if ((unsigned)xs < 256u)
                        acc += x[((b * 3 + ci) << 16) + (ys << 8) + xs] *
                               wl[c * 27 + ci * 9 + ky * 3 + kx];
                }
        }
    __builtin_nontemporal_store(acc, &out[(size_t)gp * 32 + c]);
}

// ---------------------------------------------------------------------------
// Implicit-GEMM conv via WMMA bf16, async double-buffered LDS staging.
// in: padded NHWC bf16 (258x258), wp: [Cout][K] bf16, out: NHWC f32.
// M-tile 64 pixels, N-tile 64 channels, K-slab 32 (slab s -> K = s*32).
// ---------------------------------------------------------------------------
__global__ __launch_bounds__(256)
void wmma_conv_kernel(const __hip_bfloat16* __restrict__ inP,
                      const __hip_bfloat16* __restrict__ wp,
                      const float* __restrict__ bias,
                      float* __restrict__ out,
                      int cinShift, int Cout, int Ktot) {
    __shared__ alignas(16) __hip_bfloat16 As[2][64 * 40];
    __shared__ alignas(16) __hip_bfloat16 Bs[2][64 * 40];

    const int Cin  = 1 << cinShift;
    const int cdiv = cinShift - 5;           // slabs per tap = Cin/32
    const int S    = 9 << cdiv;

    int t    = threadIdx.x;
    int lane = t & 31;
    int wave = t >> 5;
    int wm = wave & 3, wn = wave >> 2;
    int gp0 = blockIdx.x * 64;
    int n0  = blockIdx.y * 64;

    // per-thread A async-load source (16B each, 256 threads = 64px x 32k slab)
    int ap  = t >> 2;
    int akc = (t & 3) * 8;
    int gp = gp0 + ap;
    int b = gp >> 16, rem = gp & 65535, y = rem >> 8, xx = rem & 255;
    const __hip_bfloat16* abase =
        inP + (((size_t)((b * PW + y + 1) * PW + xx + 1)) << cinShift) + akc;

    // per-thread B async-load source (slab s -> wrow + s*32)
    int bn  = t & 63;
    int bkc = (t >> 6) * 8;
    const __hip_bfloat16* brow = wp + (size_t)(n0 + bn) * Ktot + bkc;

    // LDS destinations (byte offsets, per-lane)
    unsigned ldsA0 = (unsigned)(uintptr_t)&As[0][ap * 40 + akc];
    unsigned ldsA1 = (unsigned)(uintptr_t)&As[1][ap * 40 + akc];
    unsigned ldsB0 = (unsigned)(uintptr_t)&Bs[0][bn * 40 + bkc];
    unsigned ldsB1 = (unsigned)(uintptr_t)&Bs[1][bn * 40 + bkc];

    auto issue = [&](int s, int p) {
        int tap = s >> cdiv;
        int cb  = (s - (tap << cdiv)) << 5;
        int ty  = (tap * 11) >> 5;           // tap/3 for tap<=8
        int tx  = tap - ty * 3;
        const __hip_bfloat16* ga = abase + (ptrdiff_t)((ty - 1) * PW + (tx - 1)) * Cin + cb;
        const __hip_bfloat16* gb = brow + (size_t)s * 32;
        unsigned la = p ? ldsA1 : ldsA0;
        unsigned lb = p ? ldsB1 : ldsB0;
        asm volatile("global_load_async_to_lds_b128 %0, %1, off"
                     :: "v"(la), "v"(ga) : "memory");
        asm volatile("global_load_async_to_lds_b128 %0, %1, off"
                     :: "v"(lb), "v"(gb) : "memory");
    };

    v8f acc0 = {};
    v8f acc1 = {};

    // fragment LDS coordinates (per ISA VGPR layouts)
    int arow = wm * 16 + (lane & 15);
    int akb  = (lane >> 4) * 8;
    int bkb  = (lane >> 4) * 8;
    int bc0  = wn * 32 + (lane & 15);
    int bc1  = bc0 + 16;

    issue(0, 0);
    for (int s = 0; s < S; s++) {
        int p = s & 1;
        asm volatile("s_wait_asynccnt 0x0" ::: "memory");
        __syncthreads();                      // slab s visible to all waves
        if (s + 1 < S) issue(s + 1, p ^ 1);   // overlap next slab with compute

        v16bf afrag, bfrag0, bfrag1;
        ((uint4*)&afrag)[0]  = *(const uint4*)(&As[p][arow * 40 + akb]);
        ((uint4*)&afrag)[1]  = *(const uint4*)(&As[p][arow * 40 + akb + 16]);
        ((uint4*)&bfrag0)[0] = *(const uint4*)(&Bs[p][bc0 * 40 + bkb]);
        ((uint4*)&bfrag0)[1] = *(const uint4*)(&Bs[p][bc0 * 40 + bkb + 16]);
        ((uint4*)&bfrag1)[0] = *(const uint4*)(&Bs[p][bc1 * 40 + bkb]);
        ((uint4*)&bfrag1)[1] = *(const uint4*)(&Bs[p][bc1 * 40 + bkb + 16]);

        acc0 = __builtin_amdgcn_wmma_f32_16x16x32_bf16(false, afrag, false, bfrag0,
                                                       (short)0, acc0, false, false);
        acc1 = __builtin_amdgcn_wmma_f32_16x16x32_bf16(false, afrag, false, bfrag1,
                                                       (short)0, acc1, false, false);
    }

    // epilogue: C layout (VGPR r: lanes0-15 M=r, lanes16-31 M=r+8), NT stores
    int mbase = gp0 + wm * 16 + ((lane >> 4) * 8);
    int ncol  = n0 + wn * 32 + (lane & 15);
    float bs0 = bias[ncol], bs1 = bias[ncol + 16];
    for (int r = 0; r < 8; r++) {
        size_t row = (size_t)(mbase + r) * Cout;
        __builtin_nontemporal_store(acc0[r] + bs0, &out[row + ncol]);
        __builtin_nontemporal_store(acc1[r] + bs1, &out[row + ncol + 16]);
    }
}

// ---------------------------------------------------------------------------
// BN batch statistics: per-block partial sums (fixed order, no atomics)
// ---------------------------------------------------------------------------
__global__ __launch_bounds__(256)
void bn_partial_kernel(const float* __restrict__ conv, float* __restrict__ partial,
                       int Cout, int coutShift) {
    __shared__ float ssum[256];
    __shared__ float ssq[256];
    int t = threadIdx.x;
    int c  = t & (Cout - 1);
    int pi = t >> coutShift;
    int lanesPerC = 256 >> coutShift;
    const int chunk = NPIX / 256;
    int p0 = blockIdx.x * chunk;
    float s = 0.f, q = 0.f;
    for (int p = pi; p < chunk; p += lanesPerC) {
        float v = __builtin_nontemporal_load(&conv[(size_t)(p0 + p) * Cout + c]);
        s += v; q += v * v;
    }
    ssum[t] = s; ssq[t] = q;
    __syncthreads();
    for (int st = lanesPerC >> 1; st > 0; st >>= 1) {
        if (pi < st) { ssum[t] += ssum[t + st * Cout]; ssq[t] += ssq[t + st * Cout]; }
        __syncthreads();
    }
    if (pi == 0) {
        partial[blockIdx.x * 2 * Cout + c]        = ssum[t];
        partial[blockIdx.x * 2 * Cout + Cout + c] = ssq[t];
    }
}

__global__ void bn_final_kernel(const float* __restrict__ partial,
                                const float* __restrict__ gamma,
                                const float* __restrict__ beta,
                                float* __restrict__ scsh, int Cout) {
    int c = blockIdx.x * 64 + threadIdx.x;
    if (c >= Cout) return;
    float s = 0.f, q = 0.f;
    for (int i = 0; i < 256; i++) {
        s += partial[i * 2 * Cout + c];
        q += partial[i * 2 * Cout + Cout + c];
    }
    const float inv = 1.f / (float)NPIX;
    float mean = s * inv;
    float var  = q * inv - mean * mean;
    float sc = gamma[c] * rsqrtf(var + 1e-5f);
    scsh[c]        = sc;
    scsh[Cout + c] = beta[c] - mean * sc;
}

// ---------------------------------------------------------------------------
// BN affine + LeakyReLU -> padded NHWC bf16 (next stage input)
// ---------------------------------------------------------------------------
__global__ void normact_kernel(const float* __restrict__ conv, const float* __restrict__ scsh,
                               __hip_bfloat16* __restrict__ outb, int coutShift) {
    int Cout = 1 << coutShift;
    size_t i = (size_t)blockIdx.x * 256 + threadIdx.x;
    if (i >= ((size_t)NPIX << coutShift)) return;
    int c = (int)(i & (size_t)(Cout - 1));
    size_t pix = i >> coutShift;
    int b = (int)(pix >> 16), y = (int)((pix >> 8) & 255), x = (int)(pix & 255);
    float v = __builtin_nontemporal_load(&conv[i]) * scsh[c] + scsh[Cout + c];
    v = v > 0.f ? v : 0.01f * v;
    outb[(((size_t)((b * PW + y + 1) * PW + x + 1)) << coutShift) + c] = __float2bfloat16(v);
}

// ---------------------------------------------------------------------------
// Stage 4 finalize: BN+LeakyReLU + NHWC->NCHW transpose into z (f32, d_out)
// ---------------------------------------------------------------------------
__global__ __launch_bounds__(256)
void finalize_z_kernel(const float* __restrict__ conv, const float* __restrict__ scsh,
                       float* __restrict__ z) {
    __shared__ float T[32][65];
    int t = threadIdx.x;
    int pix0 = blockIdx.x * 64;
    int c0   = blockIdx.y * 32;
    {
        int cc = t & 31, pi = t >> 5;
        float sc = scsh[c0 + cc], sh = scsh[256 + c0 + cc];
        for (int i = 0; i < 8; i++) {
            int p = pi + i * 8;
            float v = __builtin_nontemporal_load(&conv[(size_t)(pix0 + p) * 256 + c0 + cc]) * sc + sh;
            T[cc][p] = v > 0.f ? v : 0.01f * v;
        }
    }
    __syncthreads();
    {
        int pp = t & 63, ci = t >> 6;
        int b = pix0 >> 16, rem = pix0 & 65535;
        for (int i = 0; i < 8; i++) {
            int c = c0 + ci + i * 4;
            __builtin_nontemporal_store(T[ci + i * 4][pp],
                                        &z[((size_t)(b * 256 + c) << 16) + rem + pp]);
        }
    }
}

// ---------------------------------------------------------------------------
// Gathers
// ---------------------------------------------------------------------------
__global__ void gather_patches_kernel(const float* __restrict__ x, const int* __restrict__ anc,
                                      float* __restrict__ out) {
    int i = blockIdx.x * 256 + threadIdx.x;       // 62208
    if (i >= 62208) return;
    int b = i / 7776;  int r  = i - b * 7776;
    int s = r / 243;   int r2 = r - s * 243;
    int c = r2 / 81;   int r3 = r2 - c * 81;
    int py = r3 / 9,   px = r3 - py * 9;
    int h = anc[(b * 32 + s) * 2];
    int w = anc[(b * 32 + s) * 2 + 1];
    out[i] = x[((b * 3 + c) * 256 + (h - 4 + py)) * 256 + (w - 4 + px)];
}

__global__ void gather_z_kernel(const float* __restrict__ z, const int* __restrict__ anc,
                                float* __restrict__ out) {
    int i = blockIdx.x * 256 + threadIdx.x;       // 65536
    if (i >= 65536) return;
    int l  = i & 255;
    int bs = i >> 8;
    int b  = bs >> 5;
    int h = anc[bs * 2], w = anc[bs * 2 + 1];
    out[i] = z[((size_t)(b * 256 + l) << 16) + h * 256 + w];
}

// ---------------------------------------------------------------------------
// Recon GEMM: x_recon[8,7776] = z_anchors[8,8192] @ rw[7776,8192]^T + rb
// Exact f32 via V_WMMA_F32_16X16X4_F32.
// ---------------------------------------------------------------------------
__global__ __launch_bounds__(128)
void recon_wmma_kernel(const float* __restrict__ za, const float* __restrict__ rw,
                       const float* __restrict__ rb, float* __restrict__ out) {
    int wave = blockIdx.x * 4 + (threadIdx.x >> 5);
    int lane = threadIdx.x & 31;
    if (wave >= 486) return;
    int o0  = wave * 16;
    int m   = lane & 15;
    int n   = lane & 15;
    int khi = (lane >> 4) * 2;
    const float* arow = za + (size_t)m * 8192;
    const float* brow = rw + (size_t)(o0 + n) * 8192;
    bool mok = m < 8;
    v8f acc = {};
    for (int k0 = 0; k0 < 8192; k0 += 4) {
        int kk = k0 + khi;
        v2f a, bf;
        a.x = mok ? arow[kk]     : 0.f;
        a.y = mok ? arow[kk + 1] : 0.f;
        bf.x = brow[kk];
        bf.y = brow[kk + 1];
        acc = __builtin_amdgcn_wmma_f32_16x16x4_f32(false, a, false, bf,
                                                    (short)0, acc, false, false);
    }
    if (lane < 16) {
        float bias = rb[o0 + n];
        for (int r = 0; r < 8; r++)
            out[(size_t)r * 7776 + o0 + n] = acc[r] + bias;
    }
}

// ---------------------------------------------------------------------------
// Host orchestration
// ---------------------------------------------------------------------------
extern "C" void kernel_launch(void* const* d_in, const int* in_sizes, int n_in,
                              void* d_out, int out_size, void* d_ws, size_t ws_size,
                              hipStream_t stream) {
    (void)in_sizes; (void)n_in; (void)out_size; (void)ws_size;
    const float* x    = (const float*)d_in[0];
    const int*   anch = (const int*)d_in[1];
    const int*   posi = (const int*)d_in[2];
    const float* w1 = (const float*)d_in[3],  *b1 = (const float*)d_in[4];
    const float* g1 = (const float*)d_in[5],  *be1 = (const float*)d_in[6];
    const float* w2 = (const float*)d_in[7],  *b2 = (const float*)d_in[8];
    const float* g2 = (const float*)d_in[9],  *be2 = (const float*)d_in[10];
    const float* w3 = (const float*)d_in[11], *b3 = (const float*)d_in[12];
    const float* g3 = (const float*)d_in[13], *be3 = (const float*)d_in[14];
    const float* w4 = (const float*)d_in[15], *b4 = (const float*)d_in[16];
    const float* g4 = (const float*)d_in[17], *be4 = (const float*)d_in[18];
    const float* rw = (const float*)d_in[19], *rb = (const float*)d_in[20];

    // d_out layout: z | x_anchors_flat | x_recon | z_anchors | z_positives
    float* z   = (float*)d_out;
    float* xa  = z  + (size_t)8 * 256 * IMG_HW;
    float* xr  = xa + 62208;
    float* zan = xr + 62208;
    float* zpo = zan + 65536;

    // workspace carve-up (256B aligned)
    char* ws = (char*)d_ws;
    size_t off = 0;
    auto take = [&](size_t bytes) { void* p = ws + off; off += (bytes + 255) & ~(size_t)255; return p; };
    float*          convbuf = (float*)take((size_t)NPIX * 256 * 4);              // 537 MB
    __hip_bfloat16* actA    = (__hip_bfloat16*)take((size_t)8 * PIMG * 128 * 2); // 136 MB (padded)
    __hip_bfloat16* actB    = (__hip_bfloat16*)take((size_t)8 * PIMG * 64 * 2);  //  68 MB (padded)
    __hip_bfloat16* wp2     = (__hip_bfloat16*)take((size_t)64 * 288 * 2);
    __hip_bfloat16* wp3     = (__hip_bfloat16*)take((size_t)128 * 576 * 2);
    __hip_bfloat16* wp4     = (__hip_bfloat16*)take((size_t)256 * 1152 * 2);
    float*          partial = (float*)take((size_t)256 * 2 * 256 * 4);
    float*          scsh    = (float*)take((size_t)2 * 256 * 4);

    // pack weights
    pack_w_kernel<<<(64 * 288 + 255) / 256, 256, 0, stream>>>(w2, wp2, 32, 64);
    pack_w_kernel<<<(128 * 576 + 255) / 256, 256, 0, stream>>>(w3, wp3, 64, 128);
    pack_w_kernel<<<(256 * 1152 + 255) / 256, 256, 0, stream>>>(w4, wp4, 128, 256);

    // stage 1: direct conv + BN + act (-> padded actA, 32ch)
    conv1_kernel<<<NPIX / 8, 256, 0, stream>>>(x, w1, b1, convbuf);
    bn_partial_kernel<<<256, 256, 0, stream>>>(convbuf, partial, 32, 5);
    bn_final_kernel<<<1, 64, 0, stream>>>(partial, g1, be1, scsh, 32);
    border_zero_kernel<<<(8 * 1028 * 32 + 255) / 256, 256, 0, stream>>>(actA, 5);
    normact_kernel<<<(unsigned)((size_t)NPIX * 32 / 256), 256, 0, stream>>>(convbuf, scsh, actA, 5);

    // stage 2: 32 -> 64
    wmma_conv_kernel<<<dim3(NPIX / 64, 1), 256, 0, stream>>>(actA, wp2, b2, convbuf, 5, 64, 288);
    bn_partial_kernel<<<256, 256, 0, stream>>>(convbuf, partial, 64, 6);
    bn_final_kernel<<<1, 64, 0, stream>>>(partial, g2, be2, scsh, 64);
    border_zero_kernel<<<(8 * 1028 * 64 + 255) / 256, 256, 0, stream>>>(actB, 6);
    normact_kernel<<<(unsigned)((size_t)NPIX * 64 / 256), 256, 0, stream>>>(convbuf, scsh, actB, 6);

    // stage 3: 64 -> 128
    wmma_conv_kernel<<<dim3(NPIX / 64, 2), 256, 0, stream>>>(actB, wp3, b3, convbuf, 6, 128, 576);
    bn_partial_kernel<<<256, 256, 0, stream>>>(convbuf, partial, 128, 7);
    bn_final_kernel<<<2, 64, 0, stream>>>(partial, g3, be3, scsh, 128);
    border_zero_kernel<<<(8 * 1028 * 128 + 255) / 256, 256, 0, stream>>>(actA, 7);
    normact_kernel<<<(unsigned)((size_t)NPIX * 128 / 256), 256, 0, stream>>>(convbuf, scsh, actA, 7);

    // stage 4: 128 -> 256, finalize into z (NCHW f32)
    wmma_conv_kernel<<<dim3(NPIX / 64, 4), 256, 0, stream>>>(actA, wp4, b4, convbuf, 7, 256, 1152);
    bn_partial_kernel<<<256, 256, 0, stream>>>(convbuf, partial, 256, 8);
    bn_final_kernel<<<4, 64, 0, stream>>>(partial, g4, be4, scsh, 256);
    finalize_z_kernel<<<dim3(NPIX / 64, 8), 256, 0, stream>>>(convbuf, scsh, z);

    // gathers
    gather_patches_kernel<<<(62208 + 255) / 256, 256, 0, stream>>>(x, anch, xa);
    gather_z_kernel<<<256, 256, 0, stream>>>(z, anch, zan);
    gather_z_kernel<<<256, 256, 0, stream>>>(z, posi, zpo);

    // recon GEMM
    recon_wmma_kernel<<<122, 128, 0, stream>>>(zan, rw, rb, xr);
}
